// S4Model_34565896798361
// MI455X (gfx1250) — matrix operand order repaired and checked by reading, not compiled, not run on hardware
//
#include <hip/hip_runtime.h>
#include <hip/hip_fp16.h>
#include <math.h>

// Problem constants (from reference)
#define BATCH 32
#define SEQL  4096
#define DIN   16
#define HD    128
#define NN    32                 // N2 complex states
#define NLAY  4
#define DOUT  8
#define NCH   32                 // scan chunks
#define CT    (SEQL / NCH)       // 128 steps per chunk
#define NCOL  (BATCH * NCH)      // 1024 GEMM columns per h

typedef __attribute__((ext_vector_type(16))) _Float16 v16h;
typedef __attribute__((ext_vector_type(8)))  _Float16 v8h;
typedef __attribute__((ext_vector_type(8)))  float    v8f;

// ---------------------------------------------------------------------------
// Encoder: u[b,l,h] = x[b,l,:] . enc_w[h,:] + enc_b[h]   (token-major B*L x H)
// ---------------------------------------------------------------------------
__global__ __launch_bounds__(256) void enc_kernel(
    const float* __restrict__ x, const float* __restrict__ w,
    const float* __restrict__ bvec, float* __restrict__ u) {
  int idx = blockIdx.x * 256 + threadIdx.x;
  int h   = idx & (HD - 1);
  int tok = idx >> 7;
  const float* xr = x + (size_t)tok * DIN;
  const float* wr = w + h * DIN;
  float acc = bvec[h];
#pragma unroll
  for (int d = 0; d < DIN; ++d) acc = fmaf(xr[d], wr[d], acc);
  u[idx] = acc;
}

// ---------------------------------------------------------------------------
// Convert one layer's conv weights (256x128 f32, row-major O x H) to f16
// ---------------------------------------------------------------------------
__global__ __launch_bounds__(256) void wcvt_kernel(
    const float* __restrict__ w, _Float16* __restrict__ wf) {
  int i = blockIdx.x * 256 + threadIdx.x;
  wf[i] = (_Float16)w[i];
}

// ---------------------------------------------------------------------------
// Tiled transpose: u token-major (B*L, H) f32 -> zT h-major f16,
// zT[(h*NCOL + b*NCH + c)*CT + j] = u[b, c*CT+j, h]
// ---------------------------------------------------------------------------
__global__ __launch_bounds__(256) void transpose_fwd(
    const float* __restrict__ u, _Float16* __restrict__ zT) {
  __shared__ float tile[32][33];
  int blk = blockIdx.x;            // (B*L/32)*(H/32) = 16384
  int hb  = blk & 3;
  int t0  = (blk >> 2) * 32;
  int h0  = hb * 32;
  int lane = threadIdx.x & 31;
  int row  = threadIdx.x >> 5;     // 0..7
#pragma unroll
  for (int r = 0; r < 4; ++r) {
    int tt = row + r * 8;
    tile[tt][lane] = u[(size_t)(t0 + tt) * HD + h0 + lane];
  }
  __syncthreads();
  int b  = t0 >> 12;
  int c  = (t0 >> 7) & (NCH - 1);
  int j0 = t0 & (CT - 1);
#pragma unroll
  for (int r = 0; r < 4; ++r) {
    int hl = row + r * 8;
    size_t col = (size_t)(h0 + hl) * NCOL + b * NCH + c;
    zT[col * CT + j0 + lane] = (_Float16)tile[lane][hl];
  }
}

// ---------------------------------------------------------------------------
// Build per-h WMMA operand matrices for one layer:
//  Toep[h]: 128x128 f16 lower-tri Toeplitz of k[h,0..127]
//  V[h]   : 64x128 f16, rows (2n,2n+1) = Re/Im(a^(127-j))  -> chunk states
//  W[h]   : 128x64 f16, cols (2n,2n+1) = 2Re / -2Im (C' a^(t+1)) -> injection
// ---------------------------------------------------------------------------
__global__ __launch_bounds__(256) void kbuild(
    const float* __restrict__ log_dt, const float* __restrict__ log_A_real,
    const float* __restrict__ A_imag, const float* __restrict__ C_re,
    const float* __restrict__ C_im, _Float16* __restrict__ Toep,
    _Float16* __restrict__ V, _Float16* __restrict__ W, int layer) {
  __shared__ float kvec[CT];
  __shared__ float sdre[NN], sdim[NN], scr[NN], sci[NN];
  int h = blockIdx.x;
  int tid = threadIdx.x;
  float dt = expf(log_dt[layer * HD + h]);
  if (tid < NN) {
    int n = tid;
    size_t pi_ = ((size_t)layer * HD + h) * NN + n;
    float Are = -expf(log_A_real[pi_]);
    float im  = A_imag[pi_];
    float dre = Are * dt, dim = im * dt;
    sdre[n] = dre; sdim[n] = dim;
    float m = expf(dre);
    float are_ = m * cosf(dim), aie_ = m * sinf(dim);
    float er = are_ - 1.0f, ei = aie_;
    float inv = 1.0f / (Are * Are + im * im);
    float tr = (er * Are + ei * im) * inv;
    float ti = (ei * Are - er * im) * inv;
    float Cr = C_re[pi_], Ci = C_im[pi_];
    scr[n] = Cr * tr - Ci * ti;
    sci[n] = Cr * ti + Ci * tr;
  }
  __syncthreads();
  if (tid < CT) {
    float t = (float)tid;
    float s = 0.f;
#pragma unroll
    for (int n = 0; n < NN; ++n) {
      float m = expf(sdre[n] * t);
      float pr = m * cosf(sdim[n] * t), pim = m * sinf(sdim[n] * t);
      s += scr[n] * pr - sci[n] * pim;
    }
    kvec[tid] = 2.f * s;
  }
  __syncthreads();
  _Float16* Th = Toep + (size_t)h * CT * CT;
  for (int e = tid; e < CT * CT; e += 256) {
    int t = e >> 7, j = e & (CT - 1);
    Th[e] = (_Float16)((j <= t) ? kvec[t - j] : 0.f);
  }
  _Float16* Vh = V + (size_t)h * 64 * CT;
  for (int e = tid; e < 64 * CT; e += 256) {
    int comp = e >> 7, j = e & (CT - 1);
    int n = comp >> 1;
    float p = (float)(CT - 1 - j);
    float m = expf(sdre[n] * p);
    float val = (comp & 1) ? m * sinf(sdim[n] * p) : m * cosf(sdim[n] * p);
    Vh[e] = (_Float16)val;
  }
  _Float16* Wh = W + (size_t)h * CT * 64;
  for (int e = tid; e < CT * 64; e += 256) {
    int t = e >> 6, comp = e & 63;
    int n = comp >> 1;
    float p = (float)(t + 1);
    float m = expf(sdre[n] * p);
    float pr = m * cosf(sdim[n] * p), pim = m * sinf(sdim[n] * p);
    float qr = scr[n] * pr - sci[n] * pim;
    float qi = scr[n] * pim + sci[n] * pr;
    Wh[e] = (_Float16)((comp & 1) ? (-2.f * qi) : (2.f * qr));
  }
}

// ---------------------------------------------------------------------------
// GEMM1: per h, S(64 x 1024) = V(64x128) @ zT(128x1024).  S f32, layout
// S[(h*NCOL+col)*64 + comp].  Block: 8 waves, tile 64 rows x 32 cols.
// ---------------------------------------------------------------------------
__global__ __launch_bounds__(256) void gemm_states(
    const _Float16* __restrict__ V, const _Float16* __restrict__ zT,
    float* __restrict__ S) {
  int blk = blockIdx.x;            // H * (NCOL/32) = 4096
  int h  = blk >> 5;
  int cs = (blk & 31) * 32;
  int lane = threadIdx.x & 31, wv = threadIdx.x >> 5;
  int mt = wv & 3, nt = wv >> 2;
  v8f acc;
#pragma unroll
  for (int j = 0; j < 8; ++j) acc[j] = 0.f;
  int arow  = mt * 16 + (lane & 15);
  int akoff = (lane >> 4) * 8;
  int bkoff = (lane >> 4) * 8;
  const _Float16* Ab = V + ((size_t)h * 64 + arow) * CT;
  size_t colg = (size_t)h * NCOL + cs + nt * 16 + (lane & 15);
  const _Float16* Bb = zT + colg * CT;
#pragma unroll
  for (int kk = 0; kk < 4; ++kk) {
    int k0 = kk * 32;
    v8h alo = *(const v8h*)(Ab + k0 + akoff);
    v8h ahi = *(const v8h*)(Ab + k0 + akoff + 16);
    v16h A = __builtin_shufflevector(alo, ahi, 0, 1, 2, 3, 4, 5, 6, 7, 8, 9,
                                     10, 11, 12, 13, 14, 15);
    v8h blo = *(const v8h*)(Bb + k0 + bkoff);
    v8h bhi = *(const v8h*)(Bb + k0 + bkoff + 16);
    v16h Bv = __builtin_shufflevector(blo, bhi, 0, 1, 2, 3, 4, 5, 6, 7, 8, 9,
                                      10, 11, 12, 13, 14, 15);
    acc = __builtin_amdgcn_wmma_f32_16x16x32_f16(false, A, false, Bv, (short)0,
                                                 acc, false, false);
  }
  int nlo = lane & 15, mh = lane >> 4;
  float* sp = S + ((size_t)h * NCOL + cs + nt * 16 + nlo) * 64 + mt * 16 + mh * 8;
#pragma unroll
  for (int j = 0; j < 8; ++j) sp[j] = acc[j];
}

// ---------------------------------------------------------------------------
// Phase B: 32-step prefix scan over chunks. Thread per (h,b,n).
// Reads S f32, writes Sin f16 (state entering each chunk) in same layout.
// ---------------------------------------------------------------------------
__global__ __launch_bounds__(256) void s4_phaseB(
    const float* __restrict__ log_dt, const float* __restrict__ log_A_real,
    const float* __restrict__ A_imag, const float* __restrict__ S,
    _Float16* __restrict__ Sin, int layer) {
  int id = blockIdx.x * 256 + threadIdx.x;  // H*B*NN = 131072
  int n = id & 31;
  int b = (id >> 5) & 31;
  int h = id >> 10;
  float dt  = expf(log_dt[layer * HD + h]);
  size_t pi_ = ((size_t)layer * HD + h) * NN + n;
  float Are = -expf(log_A_real[pi_]);
  float dim = A_imag[pi_] * dt;
  float dre = Are * dt;
  float m   = expf(dre * (float)CT);
  float aTr = m * cosf(dim * (float)CT);
  float aTi = m * sinf(dim * (float)CT);
  float str = 0.f, sti = 0.f;
  for (int c = 0; c < NCH; ++c) {
    size_t base = ((size_t)h * NCOL + b * NCH + c) * 64 + 2 * n;
    float lr = S[base], li = S[base + 1];
    Sin[base]     = (_Float16)str;
    Sin[base + 1] = (_Float16)sti;
    float nr = aTr * str - aTi * sti + lr;
    sti = aTr * sti + aTi * str + li;
    str = nr;
  }
}

// ---------------------------------------------------------------------------
// GEMM2: per h, Y(128x1024) = Toep(128x128)@zT(128x1024) + W(128x64)@Sin.
// K=192 in 6 WMMA steps (4 from Toep/zT, 2 from W/Sin). Y f16 h-major.
// Block: 8 waves, tile 128 rows x 16 cols (one wave per 16x16).
// ---------------------------------------------------------------------------
__global__ __launch_bounds__(256) void gemm_y(
    const _Float16* __restrict__ Toep, const _Float16* __restrict__ W,
    const _Float16* __restrict__ zT, const _Float16* __restrict__ Sin,
    _Float16* __restrict__ Y) {
  int blk = blockIdx.x;            // H * (NCOL/16) = 8192
  int h  = blk >> 6;
  int cs = (blk & 63) * 16;
  int lane = threadIdx.x & 31, mt = threadIdx.x >> 5;
  int arow  = mt * 16 + (lane & 15);
  int akoff = (lane >> 4) * 8;
  int bkoff = (lane >> 4) * 8;
  const _Float16* At = Toep + ((size_t)h * CT + arow) * CT;
  const _Float16* Aw = W + ((size_t)h * CT + arow) * 64;
  size_t colg = (size_t)h * NCOL + cs + (lane & 15);
  const _Float16* Bz = zT + colg * CT;
  const _Float16* Bs = Sin + colg * 64;
  v8f acc;
#pragma unroll
  for (int j = 0; j < 8; ++j) acc[j] = 0.f;
#pragma unroll
  for (int kk = 0; kk < 4; ++kk) {
    int k0 = kk * 32;
    v8h alo = *(const v8h*)(At + k0 + akoff);
    v8h ahi = *(const v8h*)(At + k0 + akoff + 16);
    v16h A = __builtin_shufflevector(alo, ahi, 0, 1, 2, 3, 4, 5, 6, 7, 8, 9,
                                     10, 11, 12, 13, 14, 15);
    v8h blo = *(const v8h*)(Bz + k0 + bkoff);
    v8h bhi = *(const v8h*)(Bz + k0 + bkoff + 16);
    v16h Bv = __builtin_shufflevector(blo, bhi, 0, 1, 2, 3, 4, 5, 6, 7, 8, 9,
                                      10, 11, 12, 13, 14, 15);
    acc = __builtin_amdgcn_wmma_f32_16x16x32_f16(false, A, false, Bv, (short)0,
                                                 acc, false, false);
  }
#pragma unroll
  for (int kk = 0; kk < 2; ++kk) {
    int k0 = kk * 32;
    v8h alo = *(const v8h*)(Aw + k0 + akoff);
    v8h ahi = *(const v8h*)(Aw + k0 + akoff + 16);
    v16h A = __builtin_shufflevector(alo, ahi, 0, 1, 2, 3, 4, 5, 6, 7, 8, 9,
                                     10, 11, 12, 13, 14, 15);
    v8h blo = *(const v8h*)(Bs + k0 + bkoff);
    v8h bhi = *(const v8h*)(Bs + k0 + bkoff + 16);
    v16h Bv = __builtin_shufflevector(blo, bhi, 0, 1, 2, 3, 4, 5, 6, 7, 8, 9,
                                      10, 11, 12, 13, 14, 15);
    acc = __builtin_amdgcn_wmma_f32_16x16x32_f16(false, A, false, Bv, (short)0,
                                                 acc, false, false);
  }
  int nlo = lane & 15, mh = lane >> 4;
  _Float16* yp = Y + ((size_t)h * NCOL + cs + nlo) * CT + mt * 16 + mh * 8;
#pragma unroll
  for (int j = 0; j < 8; ++j) yp[j] = (_Float16)acc[j];
}

// ---------------------------------------------------------------------------
// Transpose back + D-skip + exact GELU: Y,zT h-major f16 -> yact token-major
// ---------------------------------------------------------------------------
__global__ __launch_bounds__(256) void transpose_back(
    const _Float16* __restrict__ Y, const _Float16* __restrict__ zT,
    const float* __restrict__ Dl, _Float16* __restrict__ yact) {
  __shared__ _Float16 tile[32][34];
  int blk = blockIdx.x;            // 16384
  int hb  = blk & 3;
  int t0  = (blk >> 2) * 32;
  int h0  = hb * 32;
  int lane = threadIdx.x & 31;
  int row  = threadIdx.x >> 5;
  int b  = t0 >> 12;
  int c  = (t0 >> 7) & (NCH - 1);
  int j0 = t0 & (CT - 1);
#pragma unroll
  for (int r = 0; r < 4; ++r) {
    int hl = row + r * 8;
    size_t base = ((size_t)(h0 + hl) * NCOL + b * NCH + c) * CT + j0 + lane;
    float y = (float)Y[base] + Dl[h0 + hl] * (float)zT[base];
    float g = 0.5f * y * (1.f + erff(y * 0.70710678118654752f));
    tile[hl][lane] = (_Float16)g;
  }
  __syncthreads();
#pragma unroll
  for (int r = 0; r < 4; ++r) {
    int tt = row + r * 8;
    yact[(size_t)(t0 + tt) * HD + h0 + lane] = tile[lane][tt];
  }
}

// ---------------------------------------------------------------------------
// Fused 1x1 conv (WMMA) + bias + GLU + residual + LayerNorm (unchanged).
// ---------------------------------------------------------------------------
__global__ __launch_bounds__(256) void conv_glu_ln(
    const _Float16* __restrict__ yact, const _Float16* __restrict__ wf,
    const float* __restrict__ convb, const float* __restrict__ uin,
    const float* __restrict__ lnw, const float* __restrict__ lnb,
    float* __restrict__ uout) {
  __shared__ float vbuf[32 * 257];
  __shared__ float red[32 * 8 * 2];
  __shared__ float stats[32 * 2];
  int tid  = threadIdx.x;
  int lane = tid & 31;
  int wv   = tid >> 5;
  int tok0 = blockIdx.x * 32;
  int mt   = wv & 1;
  int ntg  = wv >> 1;
  v8f acc[4];
#pragma unroll
  for (int t = 0; t < 4; ++t)
#pragma unroll
    for (int j = 0; j < 8; ++j) acc[t][j] = 0.f;
  int mrow  = tok0 + mt * 16 + (lane & 15);
  int akoff = (lane >> 4) * 8;
  int bkoff = (lane >> 4) * 8;
  const _Float16* Abase = yact + (size_t)mrow * HD;
  __builtin_prefetch(uin + (size_t)(tok0 + (tid >> 3)) * HD, 0, 3);
  for (int kk = 0; kk < 4; ++kk) {
    int k0 = kk * 32;
    v8h alo = *(const v8h*)(Abase + k0 + akoff);
    v8h ahi = *(const v8h*)(Abase + k0 + akoff + 16);
    v16h A = __builtin_shufflevector(alo, ahi, 0, 1, 2, 3, 4, 5, 6, 7, 8, 9,
                                     10, 11, 12, 13, 14, 15);
#pragma unroll
    for (int t = 0; t < 4; ++t) {
      int nt = ntg * 4 + t;
      const _Float16* Bp =
          wf + ((size_t)(nt * 16 + (lane & 15))) * HD + k0 + bkoff;
      v8h blo = *(const v8h*)Bp;
      v8h bhi = *(const v8h*)(Bp + 16);
      v16h Bv = __builtin_shufflevector(blo, bhi, 0, 1, 2, 3, 4, 5, 6, 7, 8, 9,
                                        10, 11, 12, 13, 14, 15);
      acc[t] = __builtin_amdgcn_wmma_f32_16x16x32_f16(
          false, A, false, Bv, (short)0, acc[t], false, false);
    }
  }
  int nlo = lane & 15, mh = lane >> 4;
#pragma unroll
  for (int t = 0; t < 4; ++t) {
    int col = (ntg * 4 + t) * 16 + nlo;
#pragma unroll
    for (int j = 0; j < 8; ++j) {
      int m = mt * 16 + mh * 8 + j;
      vbuf[m * 257 + col] = acc[t][j];
    }
  }
  __syncthreads();
  int tk = tid >> 3;
  int seg = tid & 7;
  float vals[16];
  float s = 0.f, sq = 0.f;
  const float* ur = uin + (size_t)(tok0 + tk) * HD;
#pragma unroll
  for (int i = 0; i < 16; ++i) {
    int hh = seg * 16 + i;
    float a = vbuf[tk * 257 + hh] + convb[hh];
    float g = vbuf[tk * 257 + hh + 128] + convb[hh + 128];
    float y = a / (1.f + expf(-g));
    float r = y + ur[hh];
    vals[i] = r;
    s += r; sq += r * r;
  }
  red[(tk * 8 + seg) * 2]     = s;
  red[(tk * 8 + seg) * 2 + 1] = sq;
  __syncthreads();
  if (seg == 0) {
    float ts = 0.f, tq = 0.f;
#pragma unroll
    for (int k2 = 0; k2 < 8; ++k2) {
      ts += red[(tk * 8 + k2) * 2];
      tq += red[(tk * 8 + k2) * 2 + 1];
    }
    float mu  = ts * (1.f / HD);
    float var = tq * (1.f / HD) - mu * mu;
    stats[tk * 2]     = mu;
    stats[tk * 2 + 1] = rsqrtf(var + 1e-5f);
  }
  __syncthreads();
  float mu = stats[tk * 2], rs = stats[tk * 2 + 1];
  float* outp = uout + (size_t)(tok0 + tk) * HD;
#pragma unroll
  for (int i = 0; i < 16; ++i) {
    int hh = seg * 16 + i;
    outp[hh] = (vals[i] - mu) * rs * lnw[hh] + lnb[hh];
  }
}

// ---------------------------------------------------------------------------
// Mean-pool over L then tiny decoder with sigmoid. One block per batch item.
// ---------------------------------------------------------------------------
__global__ __launch_bounds__(256) void pool_decode(
    const float* __restrict__ emb, const float* __restrict__ dw,
    const float* __restrict__ db, float* __restrict__ pred) {
  __shared__ float ps[2][HD];
  __shared__ float pooled[HD];
  int b = blockIdx.x;
  int h = threadIdx.x & 127;
  int part = threadIdx.x >> 7;
  const float* e = emb + ((size_t)b * SEQL + (size_t)part * (SEQL / 2)) * HD + h;
  float s = 0.f;
  for (int l2 = 0; l2 < SEQL / 2; ++l2) s += e[(size_t)l2 * HD];
  ps[part][h] = s;
  __syncthreads();
  if (part == 0) pooled[h] = (ps[0][h] + ps[1][h]) * (1.0f / SEQL);
  __syncthreads();
  if (threadIdx.x < DOUT) {
    int o = threadIdx.x;
    float acc = db[o];
    for (int hh = 0; hh < HD; ++hh) acc = fmaf(pooled[hh], dw[o * HD + hh], acc);
    pred[b * DOUT + o] = 1.f / (1.f + expf(-acc));
  }
}

// ---------------------------------------------------------------------------
extern "C" void kernel_launch(void* const* d_in, const int* in_sizes, int n_in,
                              void* d_out, int out_size, void* d_ws,
                              size_t ws_size, hipStream_t stream) {
  const float* x          = (const float*)d_in[0];
  const float* enc_w      = (const float*)d_in[2];
  const float* enc_b      = (const float*)d_in[3];
  const float* log_dt     = (const float*)d_in[4];
  const float* log_A_real = (const float*)d_in[5];
  const float* A_imag     = (const float*)d_in[6];
  const float* C_re       = (const float*)d_in[7];
  const float* C_im       = (const float*)d_in[8];
  const float* Dv         = (const float*)d_in[9];
  const float* conv_w     = (const float*)d_in[10];
  const float* conv_b     = (const float*)d_in[11];
  const float* ln_w       = (const float*)d_in[12];
  const float* ln_b       = (const float*)d_in[13];
  const float* dec_w      = (const float*)d_in[14];
  const float* dec_b      = (const float*)d_in[15];
  (void)in_sizes; (void)n_in; (void)out_size; (void)ws_size;

  float* pred = (float*)d_out;
  float* emb  = pred + BATCH * DOUT;   // final u lands here (token-major B,L,H)

  const size_t NTOK  = (size_t)BATCH * SEQL;
  const size_t uB    = NTOK * HD * sizeof(float);          // 67.1 MB
  const size_t zTB   = NTOK * HD * sizeof(_Float16);       // 33.5 MB
  const size_t SB    = (size_t)HD * NCOL * 64 * sizeof(float);   // 33.5 MB
  const size_t SinB  = (size_t)HD * NCOL * 64 * sizeof(_Float16);// 16.8 MB
  const size_t yB    = NTOK * HD * sizeof(_Float16);       // 33.5 MB
  const size_t ToepB = (size_t)HD * CT * CT * sizeof(_Float16);  // 4.2 MB
  const size_t VB    = (size_t)HD * 64 * CT * sizeof(_Float16);  // 2.1 MB
  const size_t WB    = (size_t)HD * CT * 64 * sizeof(_Float16);  // 2.1 MB

  char* ws = (char*)d_ws;
  float*    uws  = (float*)ws;
  _Float16* zT   = (_Float16*)(ws + uB);
  float*    Sbuf = (float*)(ws + uB + zTB);
  _Float16* Ybuf = (_Float16*)(ws + uB + zTB);             // aliases Sbuf (dead)
  _Float16* Sin  = (_Float16*)(ws + uB + zTB + SB);
  _Float16* yact = (_Float16*)(ws + uB + zTB + SB + SinB);
  _Float16* Toep = (_Float16*)(ws + uB + zTB + SB + SinB + yB);
  _Float16* Vm   = (_Float16*)(ws + uB + zTB + SB + SinB + yB + ToepB);
  _Float16* Wm   = (_Float16*)(ws + uB + zTB + SB + SinB + yB + ToepB + VB);
  _Float16* wf   = (_Float16*)(ws + uB + zTB + SB + SinB + yB + ToepB + VB + WB);

  enc_kernel<<<(int)(NTOK * HD / 256), 256, 0, stream>>>(x, enc_w, enc_b, emb);

  float* uin = emb;
  float* uout = uws;
  for (int l = 0; l < NLAY; ++l) {
    transpose_fwd<<<(int)(NTOK / 32) * (HD / 32), 256, 0, stream>>>(uin, zT);
    kbuild<<<HD, 256, 0, stream>>>(log_dt, log_A_real, A_imag, C_re, C_im,
                                   Toep, Vm, Wm, l);
    gemm_states<<<HD * (NCOL / 32), 256, 0, stream>>>(Vm, zT, Sbuf);
    s4_phaseB<<<(HD * BATCH * NN) / 256, 256, 0, stream>>>(
        log_dt, log_A_real, A_imag, Sbuf, Sin, l);
    gemm_y<<<HD * (NCOL / 16), 256, 0, stream>>>(Toep, Wm, zT, Sin, Ybuf);
    transpose_back<<<(int)(NTOK / 32) * (HD / 32), 256, 0, stream>>>(
        Ybuf, zT, Dv + (size_t)l * HD, yact);
    wcvt_kernel<<<(2 * HD * HD) / 256, 256, 0, stream>>>(
        conv_w + (size_t)l * 2 * HD * HD, wf);
    conv_glu_ln<<<(int)(NTOK / 32), 256, 0, stream>>>(
        yact, wf, conv_b + (size_t)l * 2 * HD, uin, ln_w + (size_t)l * HD,
        ln_b + (size_t)l * HD, uout);
    float* t = uin; uin = uout; uout = t;   // after 4 swaps final u is in emb
  }

  pool_decode<<<BATCH, 256, 0, stream>>>(emb, dec_w, dec_b, pred);
}
